// NMT_33002528703371
// MI455X (gfx1250) — matrix operand order repaired and checked by reading, not compiled, not run on hardware
//
#include <hip/hip_runtime.h>
#include <hip/hip_bf16.h>
#include <math.h>

typedef __bf16 bf16;
typedef __bf16 v16bf __attribute__((ext_vector_type(16)));
typedef float  v8f   __attribute__((ext_vector_type(8)));

#define SEQ   64
#define BATCH 64
#define TSTEP 64
#define VDE   23262
#define EMB   300
#define EH    512
#define DH    1024

#define KENC  832     // 300 + 512 padded to mult of 32
#define KDEC  2368    // (1024+300) + 1024 padded to mult of 32
#define NGEN  23296   // 23262 padded to mult of 64

// ---------------------------------------------------------------------------
// WMMA fragment loaders (gfx1250 v_wmma_f32_16x16x32_bf16; contiguous k order used here for A and B alike)
// A 16x32 (MxK): lane r=M; half=0 -> K {0..15}, half=1 -> K {16..31}
// B 32x16 (KxN): lane r=N; half=0 -> K {0..15}, half=1 -> K {16..31}
// ---------------------------------------------------------------------------
__device__ __forceinline__ v16bf load_frag_a(const bf16* row, int k0, int half) {
  const bf16* q = row + k0 + half * 16;
  v16bf f;
#pragma unroll
  for (int e = 0; e < 16; ++e)
    f[e] = q[e];                         // K=e
  return f;
}

__device__ __forceinline__ v16bf load_frag_b(const bf16* row, int k0, int half) {
  const bf16* q = row + k0 + half * 16;  // 32-byte aligned by construction
  return *(const v16bf*)q;
}

// ---------------------------------------------------------------------------
// NT GEMM: C[64,N] = A[64,K](bf16,row-major) * W[N,K](bf16,row-major)^T + bias
// One wave (32 thr) per 16x64 tile. K mult of 32, N mult of 64 (padded).
// MODE 0: fp32 out; 1: bf16 out; 2: tanh -> bf16 out
// ---------------------------------------------------------------------------
template <int MODE>
__global__ __launch_bounds__(32) void gemm_wave(
    const bf16* __restrict__ A, int lda,
    const bf16* __restrict__ W, int ldw,
    const float* __restrict__ bias,
    void* __restrict__ Cout, int ldc, int K) {
  const int lane = threadIdx.x;
  const int r = lane & 15, half = lane >> 4;
  const int row0 = blockIdx.y * 16;
  const int col0 = blockIdx.x * 64;

  const bf16* arow = A + (size_t)(row0 + r) * lda;
  const bf16* wrow = W + (size_t)(col0 + r) * ldw;

  v8f acc0 = {}, acc1 = {}, acc2 = {}, acc3 = {};
  for (int k0 = 0; k0 < K; k0 += 32) {
    v16bf a = load_frag_a(arow, k0, half);
    v16bf b0 = load_frag_b(wrow,                     k0, half);
    v16bf b1 = load_frag_b(wrow + (size_t)16 * ldw,  k0, half);
    v16bf b2 = load_frag_b(wrow + (size_t)32 * ldw,  k0, half);
    v16bf b3 = load_frag_b(wrow + (size_t)48 * ldw,  k0, half);
    acc0 = __builtin_amdgcn_wmma_f32_16x16x32_bf16(false, a, false, b0, (short)0, acc0, false, false);
    acc1 = __builtin_amdgcn_wmma_f32_16x16x32_bf16(false, a, false, b1, (short)0, acc1, false, false);
    acc2 = __builtin_amdgcn_wmma_f32_16x16x32_bf16(false, a, false, b2, (short)0, acc2, false, false);
    acc3 = __builtin_amdgcn_wmma_f32_16x16x32_bf16(false, a, false, b3, (short)0, acc3, false, false);
  }

  v8f accs[4] = {acc0, acc1, acc2, acc3};
#pragma unroll
  for (int j = 0; j < 4; ++j) {
    const int n = col0 + j * 16 + r;
    const float bv = bias ? bias[n] : 0.0f;
#pragma unroll
    for (int v = 0; v < 8; ++v) {
      const int m = row0 + v + half * 8;   // C/D layout: VGPR v -> M = v + 8*half
      float val = accs[j][v] + bv;
      if (MODE == 0)      ((float*)Cout)[(size_t)m * ldc + n] = val;
      else if (MODE == 1) ((bf16*)Cout)[(size_t)m * ldc + n] = (bf16)val;
      else                ((bf16*)Cout)[(size_t)m * ldc + n] = (bf16)tanhf(val);
    }
  }
}

// ------------------------------ prep kernels -------------------------------
__global__ void k_wcat(const float* __restrict__ s0, int K0,
                       const float* __restrict__ s1, int K1,
                       bf16* __restrict__ dst, int dstK, int N) {
  size_t idx = (size_t)blockIdx.x * 256 + threadIdx.x;
  if (idx >= (size_t)N * dstK) return;
  int n = (int)(idx / dstK), k = (int)(idx % dstK);
  float v = 0.0f;
  if (k < K0) v = s0[(size_t)n * K0 + k];
  else if (k < K0 + K1) v = s1[(size_t)n * K1 + (k - K0)];
  dst[idx] = (bf16)v;
}

__global__ void k_wconv(const float* __restrict__ src, int srcN, int K,
                        bf16* __restrict__ dst, int dstN) {
  size_t idx = (size_t)blockIdx.x * 256 + threadIdx.x;
  if (idx >= (size_t)dstN * K) return;
  int n = (int)(idx / K);
  dst[idx] = (bf16)(n < srcN ? src[idx] : 0.0f);
}

__global__ void k_badd(const float* a, const float* b, float* d, int n) {
  int i = blockIdx.x * 256 + threadIdx.x;
  if (i < n) d[i] = a[i] + b[i];
}

__global__ void k_bpad(const float* s, float* d, int n, int ntot) {
  int i = blockIdx.x * 256 + threadIdx.x;
  if (i < ntot) d[i] = (i < n) ? s[i] : 0.0f;
}

__global__ void k_zero_bf(bf16* p, int n) {
  int i = blockIdx.x * 256 + threadIdx.x;
  if (i < n) p[i] = (bf16)0.0f;
}

__global__ void k_zero_f(float* p, int n) {
  int i = blockIdx.x * 256 + threadIdx.x;
  if (i < n) p[i] = 0.0f;
}

// --------------------------- encoder step kernels --------------------------
__global__ void k_xin_enc(const int* __restrict__ src, int s,
                          const float* __restrict__ emb,
                          const bf16* __restrict__ hbf,
                          bf16* __restrict__ xin) {
  int idx = blockIdx.x * 256 + threadIdx.x;          // b*KENC + k
  if (idx >= BATCH * KENC) return;
  int b = idx / KENC, k = idx % KENC;
  bf16 v;
  if (k < EMB)            v = (bf16)emb[(size_t)src[s * BATCH + b] * EMB + k];
  else if (k < EMB + EH)  v = hbf[b * EH + (k - EMB)];
  else                    v = (bf16)0.0f;
  xin[idx] = v;
}

// gate order i,f,g,o ; H = hidden size; hs_out may be null (decoder)
__global__ void k_lstm_cell(const float* __restrict__ g, float* __restrict__ c,
                            bf16* __restrict__ hbf, bf16* __restrict__ hs_out,
                            int H) {
  int idx = blockIdx.x * 256 + threadIdx.x;          // b*H + n
  if (idx >= BATCH * H) return;
  int b = idx / H, n = idx % H;
  const float* gr = g + (size_t)b * 4 * H;
  float gi = 1.0f / (1.0f + expf(-gr[n]));
  float gf = 1.0f / (1.0f + expf(-gr[H + n]));
  float gg = tanhf(gr[2 * H + n]);
  float go = 1.0f / (1.0f + expf(-gr[3 * H + n]));
  float cn = gf * c[idx] + gi * gg;
  c[idx] = cn;
  float h = go * tanhf(cn);
  hbf[idx] = (bf16)h;
  if (hs_out) hs_out[(size_t)b * (2 * EH) + n] = (bf16)h;
}

// --------------------------- decoder step kernels --------------------------
__global__ void k_dec_init(const float* __restrict__ ht0,
                           const float* __restrict__ cf,
                           const float* __restrict__ cr,
                           bf16* __restrict__ hbf, float* __restrict__ c) {
  int idx = blockIdx.x * 256 + threadIdx.x;          // b*DH + d
  if (idx >= BATCH * DH) return;
  int b = idx / DH, d = idx % DH;
  hbf[idx] = (bf16)ht0[d];
  c[idx] = (d & 1) ? cr[b * EH + (d >> 1)] : cf[b * EH + (d >> 1)];
}

// scores e[s,b] = <hs[s,b,:], sp[b,:]> ; softmax over b (reference axis=1)
__global__ void k_attn_scores(const bf16* __restrict__ hs,
                              const bf16* __restrict__ sp,
                              float* __restrict__ a) {
  int s = blockIdx.x;
  int b = threadIdx.x;                                // 64 threads
  const bf16* hrow = hs + ((size_t)s * BATCH + b) * (2 * EH);
  const bf16* srow = sp + (size_t)b * DH;
  float acc = 0.0f;
  for (int d = 0; d < 2 * EH; ++d) acc += (float)hrow[d] * (float)srow[d];
  __shared__ float sv[64];
  sv[b] = acc; __syncthreads();
  for (int off = 32; off > 0; off >>= 1) {
    if (b < off) sv[b] = fmaxf(sv[b], sv[b + off]);
    __syncthreads();
  }
  float mx = sv[0]; __syncthreads();
  float ex = expf(acc - mx);
  sv[b] = ex; __syncthreads();
  for (int off = 32; off > 0; off >>= 1) {
    if (b < off) sv[b] += sv[b + off];
    __syncthreads();
  }
  a[s * BATCH + b] = ex / sv[0];
}

// s_t[b,:] = sum_s a[s,b]*hs[s,b,:] ; cat = [s_t | h]
__global__ void k_attn_ctx(const float* __restrict__ a,
                           const bf16* __restrict__ hs,
                           const bf16* __restrict__ hdec,
                           bf16* __restrict__ cat) {
  int b = blockIdx.x;
  for (int d = threadIdx.x; d < 2 * EH; d += 256) {
    float acc = 0.0f;
    for (int s = 0; s < SEQ; ++s)
      acc += a[s * BATCH + b] * (float)hs[((size_t)s * BATCH + b) * (2 * EH) + d];
    cat[(size_t)b * (2 * DH) + d] = (bf16)acc;
    cat[(size_t)b * (2 * DH) + DH + d] = hdec[(size_t)b * DH + d];
  }
}

// fill xin_dec[b, 1024:2368] = [emb_de(trg) | h | zeros] (c_t written by GEMM)
__global__ void k_xin_dec(const int* __restrict__ trg, int t,
                          const float* __restrict__ emb,
                          const bf16* __restrict__ hbf,
                          bf16* __restrict__ xin) {
  int idx = blockIdx.x * 256 + threadIdx.x;          // b*1344 + (k-1024)
  if (idx >= BATCH * (KDEC - DH)) return;
  int b = idx / (KDEC - DH), k = DH + idx % (KDEC - DH);
  bf16 v;
  if (k < DH + EMB)           v = (bf16)emb[(size_t)trg[t * BATCH + b] * EMB + (k - DH)];
  else if (k < DH + EMB + DH) v = hbf[(size_t)b * DH + (k - DH - EMB)];
  else                        v = (bf16)0.0f;
  xin[(size_t)b * KDEC + k] = v;
}

__global__ void k_logsoftmax(const float* __restrict__ logits,
                             float* __restrict__ out, int t) {
  int b = blockIdx.x;
  int tid = threadIdx.x;                              // 256 threads
  const float* row = logits + (size_t)b * NGEN;
  __shared__ float sv[256];
  float mx = -1e30f;
  for (int v = tid; v < VDE; v += 256) mx = fmaxf(mx, row[v]);
  sv[tid] = mx; __syncthreads();
  for (int off = 128; off > 0; off >>= 1) {
    if (tid < off) sv[tid] = fmaxf(sv[tid], sv[tid + off]);
    __syncthreads();
  }
  mx = sv[0]; __syncthreads();
  float sum = 0.0f;
  for (int v = tid; v < VDE; v += 256) sum += expf(row[v] - mx);
  sv[tid] = sum; __syncthreads();
  for (int off = 128; off > 0; off >>= 1) {
    if (tid < off) sv[tid] += sv[tid + off];
    __syncthreads();
  }
  float lse = mx + logf(sv[0]);
  float* orow = out + ((size_t)(t + 1) * BATCH + b) * VDE;
  for (int v = tid; v < VDE; v += 256) orow[v] = row[v] - lse;
}

// ---------------------------------------------------------------------------
extern "C" void kernel_launch(void* const* d_in, const int* in_sizes, int n_in,
                              void* d_out, int out_size, void* d_ws, size_t ws_size,
                              hipStream_t stream) {
  const int*   src    = (const int*)d_in[0];
  const int*   trg    = (const int*)d_in[1];
  const float* emb_en = (const float*)d_in[2];
  const float* emb_de = (const float*)d_in[3];
  const float* W_ih_f = (const float*)d_in[4];
  const float* W_hh_f = (const float*)d_in[5];
  const float* b_ih_f = (const float*)d_in[6];
  const float* b_hh_f = (const float*)d_in[7];
  const float* W_ih_r = (const float*)d_in[8];
  const float* W_hh_r = (const float*)d_in[9];
  const float* b_ih_r = (const float*)d_in[10];
  const float* b_hh_r = (const float*)d_in[11];
  const float* W_ih_d = (const float*)d_in[12];
  const float* W_hh_d = (const float*)d_in[13];
  const float* b_ih_d = (const float*)d_in[14];
  const float* b_hh_d = (const float*)d_in[15];
  const float* gen_W  = (const float*)d_in[16];
  const float* gen_b  = (const float*)d_in[17];
  const float* Wi_W   = (const float*)d_in[18];
  const float* Wi_b   = (const float*)d_in[19];
  const float* W_o    = (const float*)d_in[20];
  const float* ht0    = (const float*)d_in[21];
  float* out = (float*)d_out;

  // ---- workspace arena (256B aligned) ----
  char* p = (char*)d_ws;
  auto alloc = [&](size_t bytes) -> void* {
    void* r = (void*)p;
    p += (bytes + 255) & ~(size_t)255;
    return r;
  };
  bf16* Wenc_f  = (bf16*)alloc((size_t)4 * EH * KENC * 2);
  bf16* Wenc_r  = (bf16*)alloc((size_t)4 * EH * KENC * 2);
  bf16* Wdec    = (bf16*)alloc((size_t)4 * DH * KDEC * 2);
  bf16* Wi      = (bf16*)alloc((size_t)DH * DH * 2);
  bf16* Wo      = (bf16*)alloc((size_t)DH * 2 * DH * 2);
  bf16* Wgen    = (bf16*)alloc((size_t)NGEN * DH * 2);
  bf16* hs      = (bf16*)alloc((size_t)SEQ * BATCH * 2 * EH * 2);
  bf16* h_enc   = (bf16*)alloc((size_t)BATCH * EH * 2);
  bf16* h_dec   = (bf16*)alloc((size_t)BATCH * DH * 2);
  bf16* sp_bf   = (bf16*)alloc((size_t)BATCH * DH * 2);
  bf16* cat_bf  = (bf16*)alloc((size_t)BATCH * 2 * DH * 2);
  bf16* xin_e   = (bf16*)alloc((size_t)BATCH * KENC * 2);
  bf16* xin_d   = (bf16*)alloc((size_t)BATCH * KDEC * 2);
  float* bias_f = (float*)alloc((size_t)4 * EH * 4);
  float* bias_r = (float*)alloc((size_t)4 * EH * 4);
  float* bias_d = (float*)alloc((size_t)4 * DH * 4);
  float* bias_g = (float*)alloc((size_t)NGEN * 4);
  float* g_enc  = (float*)alloc((size_t)BATCH * 4 * EH * 4);
  float* g_dec  = (float*)alloc((size_t)BATCH * 4 * DH * 4);
  float* c_f    = (float*)alloc((size_t)BATCH * EH * 4);
  float* c_r    = (float*)alloc((size_t)BATCH * EH * 4);
  float* c_dec  = (float*)alloc((size_t)BATCH * DH * 4);
  float* attn_a = (float*)alloc((size_t)SEQ * BATCH * 4);
  float* logits = (float*)alloc((size_t)BATCH * NGEN * 4);
  if ((size_t)(p - (char*)d_ws) > ws_size) return;  // ws too small: bail

  auto blks = [](size_t n) { return (unsigned)((n + 255) / 256); };

  // ---- one-time (per call) weight conversion / packing ----
  k_wcat<<<blks((size_t)4 * EH * KENC), 256, 0, stream>>>(W_ih_f, EMB, W_hh_f, EH, Wenc_f, KENC, 4 * EH);
  k_wcat<<<blks((size_t)4 * EH * KENC), 256, 0, stream>>>(W_ih_r, EMB, W_hh_r, EH, Wenc_r, KENC, 4 * EH);
  k_wcat<<<blks((size_t)4 * DH * KDEC), 256, 0, stream>>>(W_ih_d, DH + EMB, W_hh_d, DH, Wdec, KDEC, 4 * DH);
  k_wconv<<<blks((size_t)DH * DH), 256, 0, stream>>>(Wi_W, DH, DH, Wi, DH);
  k_wconv<<<blks((size_t)DH * 2 * DH), 256, 0, stream>>>(W_o, DH, 2 * DH, Wo, DH);
  k_wconv<<<blks((size_t)NGEN * DH), 256, 0, stream>>>(gen_W, VDE, DH, Wgen, NGEN);
  k_badd<<<blks(4 * EH), 256, 0, stream>>>(b_ih_f, b_hh_f, bias_f, 4 * EH);
  k_badd<<<blks(4 * EH), 256, 0, stream>>>(b_ih_r, b_hh_r, bias_r, 4 * EH);
  k_badd<<<blks(4 * DH), 256, 0, stream>>>(b_ih_d, b_hh_d, bias_d, 4 * DH);
  k_bpad<<<blks(NGEN), 256, 0, stream>>>(gen_b, bias_g, VDE, NGEN);
  k_zero_f<<<blks((size_t)BATCH * VDE), 256, 0, stream>>>(out, BATCH * VDE);  // out[0] = 0

  // ---- encoder: forward then backward ----
  for (int dir = 0; dir < 2; ++dir) {
    const bf16* W = dir ? Wenc_r : Wenc_f;
    const float* bias = dir ? bias_r : bias_f;
    float* c = dir ? c_r : c_f;
    k_zero_bf<<<blks(BATCH * EH), 256, 0, stream>>>(h_enc, BATCH * EH);
    k_zero_f<<<blks(BATCH * EH), 256, 0, stream>>>(c, BATCH * EH);
    for (int j = 0; j < SEQ; ++j) {
      int s = dir ? (SEQ - 1 - j) : j;
      k_xin_enc<<<blks(BATCH * KENC), 256, 0, stream>>>(src, s, emb_en, h_enc, xin_e);
      gemm_wave<0><<<dim3(4 * EH / 64, 4), 32, 0, stream>>>(xin_e, KENC, W, KENC, bias, g_enc, 4 * EH, KENC);
      bf16* hs_out = hs + (size_t)s * BATCH * 2 * EH + (dir ? EH : 0);
      k_lstm_cell<<<blks(BATCH * EH), 256, 0, stream>>>(g_enc, c, h_enc, hs_out, EH);
    }
  }

  // ---- decoder init: h = ht0 broadcast, c = interleave(cf, cb) ----
  k_dec_init<<<blks(BATCH * DH), 256, 0, stream>>>(ht0, c_f, c_r, h_dec, c_dec);

  // ---- decoder steps ----
  for (int t = 0; t < TSTEP - 1; ++t) {
    // sp = h @ Wi_W.T + Wi_b   (bf16 out)
    gemm_wave<1><<<dim3(DH / 64, 4), 32, 0, stream>>>(h_dec, DH, Wi, DH, Wi_b, sp_bf, DH, DH);
    // attention
    k_attn_scores<<<SEQ, 64, 0, stream>>>(hs, sp_bf, attn_a);
    k_attn_ctx<<<BATCH, 256, 0, stream>>>(attn_a, hs, h_dec, cat_bf);
    // c_t = tanh([s_t|h] @ W_o.T)  -> xin_d[:, 0:1024] (bf16)
    gemm_wave<2><<<dim3(DH / 64, 4), 32, 0, stream>>>(cat_bf, 2 * DH, Wo, 2 * DH, nullptr, xin_d, KDEC, 2 * DH);
    // xin_d[:, 1024:2368] = [emb_de(trg[t]) | h | 0]
    k_xin_dec<<<blks((size_t)BATCH * (KDEC - DH)), 256, 0, stream>>>(trg, t, emb_de, h_dec, xin_d);
    // g = xin_d @ [W_ih_d|W_hh_d].T + (b_ih_d+b_hh_d)
    gemm_wave<0><<<dim3(4 * DH / 64, 4), 32, 0, stream>>>(xin_d, KDEC, Wdec, KDEC, bias_d, g_dec, 4 * DH, KDEC);
    k_lstm_cell<<<blks(BATCH * DH), 256, 0, stream>>>(g_dec, c_dec, h_dec, nullptr, DH);
    // logits = h @ gen_W.T + gen_b ; log_softmax -> out[t+1]
    gemm_wave<0><<<dim3(NGEN / 64, 4), 32, 0, stream>>>(h_dec, DH, Wgen, DH, bias_g, logits, NGEN, DH);
    k_logsoftmax<<<BATCH, 256, 0, stream>>>(logits, out, t);
  }
}
